// RNNModel_55499567399101
// MI455X (gfx1250) — matrix operand order repaired and checked by reading, not compiled, not run on hardware
//
#include <hip/hip_runtime.h>

// ---------------------------------------------------------------------------
// RNN forward for MI455X (gfx1250, wave32, WMMA).
//
//   h_{t} = tanh(x_t * W_ih + h_{t-1} @ W_hh^T + b),  out = h_T @ fc_w^T + fc_b
//
// One wave owns 16 batch rows for all 512 steps. Per step compute the
// transposed product  h_new^T = W_hh @ h^T  with v_wmma_f32_16x16x32_f16:
//   A = W_hh   (16x32 f16 tiles, weight-stationary in VGPRs, loaded once)
//   B = h^T    (32x16 f16, batch = N dimension)
//   C = x_t*W_ih + bias   (folded into the accumulator for free)
// The WMMA D layout (lane = batch col, VGPR = hidden row) converts to the
// next step's B layout with just a half-wave lane swap: ds_swizzle SWAPX16.
// No LDS memory, no barriers, hidden state never leaves registers.
// x is streamed one float4 (4 timesteps) at a time, software-pipelined one
// iteration ahead so the global load never stalls the serial recurrence.
// ---------------------------------------------------------------------------

typedef __attribute__((ext_vector_type(16))) _Float16 v16h;
typedef __attribute__((ext_vector_type(8)))  float    v8f;
typedef __attribute__((ext_vector_type(4)))  float    v4f;
typedef __attribute__((ext_vector_type(8)))  unsigned v8u;

#define RNN_H 64
#define RNN_T 512
#define SWZ_SWAPX16 0x401F   // group-of-32: xor=0x10, or=0, and=0x1f -> lane ^ 16

static __device__ __forceinline__ unsigned pk2(float a, float b) {
  auto p = __builtin_amdgcn_cvt_pkrtz(a, b);        // f16x2 pack
  return __builtin_bit_cast(unsigned, p);
}

static __device__ __forceinline__ float fast_tanh(float x) {
#if __has_builtin(__builtin_amdgcn_tanhf)
  return __builtin_amdgcn_tanhf(x);                 // v_tanh_f32 (gfx1250 TRANS)
#elif __has_builtin(__builtin_amdgcn_tanh_f32)
  return __builtin_amdgcn_tanh_f32(x);
#else
  // tanh(x) = 1 - 2/(exp(2x)+1), exp via v_exp_f32 (2*log2(e) premul)
  float e = __builtin_amdgcn_exp2f(x * 2.88539008177793f);
  return 1.0f - 2.0f * __builtin_amdgcn_rcpf(e + 1.0f);
#endif
}

static __device__ __forceinline__ unsigned swap16(unsigned v) {
  return (unsigned)__builtin_amdgcn_ds_swizzle((int)v, SWZ_SWAPX16);
}

__global__ __launch_bounds__(32) void rnn_wmma_kernel(
    const float* __restrict__ x,      // [B, T]  (I == 1)
    const float* __restrict__ W_ih,   // [H]
    const float* __restrict__ W_hh,   // [H, H] row-major
    const float* __restrict__ b_ih,   // [H]
    const float* __restrict__ b_hh,   // [H]
    const float* __restrict__ fc_w,   // [H]
    const float* __restrict__ fc_b,   // [1]
    float* __restrict__ out)          // [B]
{
  const int lane = (int)threadIdx.x;  // 0..31
  const int j    = lane & 15;         // batch column within tile
  const int hi   = lane >> 4;         // half-wave id
  const int bj   = (int)blockIdx.x * 16 + j;   // this lane's batch row

  // ---- W_hh into A-layout f16 registers (weight stationary) --------------
  // Tile (it, kc): rows n = it*16 + j ; lane holds
  //   u[0..3]: K = kc*32 + hi*16 + {0..7},  u[4..7]: K = kc*32 + hi*16 + 8 + {0..7}
  v16h A[4][2];
#pragma unroll
  for (int it = 0; it < 4; ++it) {
    const float* wrow = W_hh + (it * 16 + j) * RNN_H;
#pragma unroll
    for (int kc = 0; kc < 2; ++kc) {
      const int k0 = kc * 32 + hi * 16;
      const int k1 = kc * 32 + hi * 16 + 8;
      v8u a;
#pragma unroll
      for (int p = 0; p < 4; ++p) {
        a[p]     = pk2(wrow[k0 + 2 * p], wrow[k0 + 2 * p + 1]);
        a[p + 4] = pk2(wrow[k1 + 2 * p], wrow[k1 + 2 * p + 1]);
      }
      A[it][kc] = __builtin_bit_cast(v16h, a);
    }
  }

  // Per-lane elementwise constants in D' layout: i(it,v) = it*16 + hi*8 + v
  float w8[4][8], bb8[4][8];
#pragma unroll
  for (int it = 0; it < 4; ++it)
#pragma unroll
    for (int v = 0; v < 8; ++v) {
      const int i = it * 16 + hi * 8 + v;
      w8[it][v]  = W_ih[i];
      bb8[it][v] = b_ih[i] + b_hh[i];
    }

  // Hidden state h^T in B layout (f16). h0 = 0.
  v16h Bh[2];
  Bh[0] = (v16h){};
  Bh[1] = (v16h){};

  const float* xrow = x + (size_t)bj * RNN_T;

  // ------------------------------ time loop -------------------------------
  // Software pipeline: xnext holds the float4 for the *next* t0 iteration,
  // loaded ~150 instructions before use so the load latency is fully hidden.
  v4f xnext = *(const v4f*)(xrow);                 // 16B-aligned (T=512)
  for (int t0 = 0; t0 < RNN_T; t0 += 4) {
    const v4f xv = xnext;
    const int tn = (t0 + 4 < RNN_T) ? (t0 + 4) : 0;  // clamp: stay in-bounds
    xnext = *(const v4f*)(xrow + tn);
#pragma unroll
    for (int u = 0; u < 4; ++u) {
      const float xs = xv[u];

      // D' tiles: C seeded with x*W_ih + bias, then accumulate W_hh @ h^T
      unsigned pk[4][4];
#pragma unroll
      for (int it = 0; it < 4; ++it) {
        v8f c;
#pragma unroll
        for (int v = 0; v < 8; ++v) c[v] = __builtin_fmaf(xs, w8[it][v], bb8[it][v]);
        c = __builtin_amdgcn_wmma_f32_16x16x32_f16(false, A[it][0], false, Bh[0],
                                                   (short)0, c, false, false);
        c = __builtin_amdgcn_wmma_f32_16x16x32_f16(false, A[it][1], false, Bh[1],
                                                   (short)0, c, false, false);
        // tanh + pack pairs (hidden rows 2p, 2p+1) to f16x2
#pragma unroll
        for (int p = 0; p < 4; ++p)
          pk[it][p] = pk2(fast_tanh(c[2 * p]), fast_tanh(c[2 * p + 1]));
      }

      // D' -> B relayout: half-wave swap via ds_swizzle (no LDS memory).
      //   hi==0 lane: keep tile 2kc,  receive partner's tile 2kc   (send 2kc+1)
      //   hi==1 lane: keep tile 2kc+1, receive partner's tile 2kc+1 (send 2kc)
#pragma unroll
      for (int kc = 0; kc < 2; ++kc) {
        v8u b;
#pragma unroll
        for (int p = 0; p < 4; ++p) {
          const unsigned own_e = pk[2 * kc][p];
          const unsigned own_o = pk[2 * kc + 1][p];
          const unsigned send  = (hi == 0) ? own_o : own_e;
          const unsigned recv  = swap16(send);
          b[p]     = (hi == 0) ? own_e : recv;   // K = kc*32 + hi*16 + {0..7}
          b[p + 4] = (hi == 0) ? recv  : own_o;  // K = kc*32 + hi*16 + {8..15}
        }
        Bh[kc] = __builtin_bit_cast(v16h, b);
      }
    }
  }

  // ------------------------- final linear layer ---------------------------
  // Lane (j,hi) holds h[bj][k] for k = kc*32 + hi*16 + s (32 of 64 values);
  // partner lane holds the other 32. Dot with fc_w, swap-reduce, store.
  float part = 0.0f;
#pragma unroll
  for (int kc = 0; kc < 2; ++kc)
#pragma unroll
    for (int s = 0; s < 16; ++s) {
      const int k = kc * 32 + hi * 16 + s;
      part = __builtin_fmaf((float)Bh[kc][s], fc_w[k], part);
    }
  part += __builtin_bit_cast(float, (int)swap16(__builtin_bit_cast(unsigned, part)));
  if (hi == 0) out[bj] = part + fc_b[0];
}

// ---------------------------------------------------------------------------
extern "C" void kernel_launch(void* const* d_in, const int* in_sizes, int n_in,
                              void* d_out, int out_size, void* d_ws, size_t ws_size,
                              hipStream_t stream) {
  (void)in_sizes; (void)n_in; (void)d_ws; (void)ws_size;
  const float* x    = (const float*)d_in[0];
  const float* W_ih = (const float*)d_in[1];
  const float* W_hh = (const float*)d_in[2];
  const float* b_ih = (const float*)d_in[3];
  const float* b_hh = (const float*)d_in[4];
  const float* fc_w = (const float*)d_in[5];
  const float* fc_b = (const float*)d_in[6];
  float* out = (float*)d_out;

  const int B = 4096;                 // out_size == B
  (void)out_size;
  dim3 grid(B / 16), block(32);       // 256 independent single-wave blocks
  rnn_wmma_kernel<<<grid, block, 0, stream>>>(x, W_ih, W_hh, b_ih, b_hh,
                                              fc_w, fc_b, out);
}
